// NystromformerEncoder_412316860612
// MI455X (gfx1250) — compile-verified, matrix operand order checked by reading, not run on hardware
//
#include <hip/hip_runtime.h>
#include <math.h>

typedef _Float16 half_t;
typedef __attribute__((ext_vector_type(16))) _Float16 v16h;
typedef __attribute__((ext_vector_type(8)))  float    v8f;

constexpr int Bb = 8, Ee = 1024, Ss = 2000, Hh = 16, HD = 64, Mm = 10, MLPd = 4096;
constexpr int SM   = Ss / Mm;     // 200
constexpr int QKVN = 3 * Ee;      // 3072
constexpr float SCALE_QK = 0.35355339059327373f;  // 64^-0.25
constexpr float LNEPS = 1e-5f;

// ---------------------------------------------------------------------------
// Elementwise / conversion kernels
// ---------------------------------------------------------------------------
__global__ void xpose_to_f16_kernel(const float* __restrict__ X, half_t* __restrict__ xh) {
  int idx = blockIdx.x * blockDim.x + threadIdx.x;   // (b,s,e) row-major
  if (idx >= Bb * Ss * Ee) return;
  int e = idx % Ee;
  int rem = idx / Ee;
  int s = rem % Ss;
  int b = rem / Ss;
  xh[idx] = (half_t)X[((size_t)b * Ee + e) * Ss + s];
}

__global__ void cvt_f16_kernel(const float* __restrict__ src, half_t* __restrict__ dst, int n) {
  int i = blockIdx.x * blockDim.x + threadIdx.x;
  if (i < n) dst[i] = (half_t)src[i];
}

__global__ void build_bias_scale_kernel(const float* __restrict__ bq, const float* __restrict__ bk,
                                        const float* __restrict__ bvv,
                                        float* __restrict__ biasc, float* __restrict__ scalec) {
  int c = blockIdx.x * blockDim.x + threadIdx.x;
  if (c >= QKVN) return;
  int which = c >> 10;
  int rem = c & 1023;
  float bb;
  if (which == 0) bb = bq[rem];
  else if (which == 1) bb = bk[rem];
  else bb = bvv[rem];
  biasc[c] = bb;
  scalec[c] = (which < 2) ? SCALE_QK : 1.0f;
}

// ---------------------------------------------------------------------------
// CDNA5 async global->LDS copy (ASYNCcnt tracked), 16B per lane.
// LDS operand is the low 32 bits of the generic pointer (flat->LDS aperture
// mapping uses addr[31:0]).
// ---------------------------------------------------------------------------
__device__ __forceinline__ void async_copy_b128(const half_t* lds_dst, const half_t* gsrc) {
  uint32_t lds = (uint32_t)(uintptr_t)lds_dst;
  uint64_t ga  = (uint64_t)(uintptr_t)gsrc;
  asm volatile("global_load_async_to_lds_b128 %0, %1, off"
               :: "v"(lds), "v"(ga)
               : "memory");
}

// ---------------------------------------------------------------------------
// WMMA tiled GEMM:  C[M x N] = A[M x K] * B[N x K]^T   (f16 in, f32 acc)
// MODE 1: Cf = (acc + bias[n]) * scalev[n]      (QKV projection)
// MODE 2: Ch = gelu_exact(acc + bias[n])        (MLP up, f16 out)
// MODE 3: Cf = acc + bias[n]                    (MLP down)
// Block: 128 threads = 4 wave32, 64x64 tile, BK = 32.
// Double-buffered LDS staging via GLOBAL_LOAD_ASYNC_TO_LDS_B128:
//   each thread issues 4 async b128 copies per K-tile (2 for A, 2 for B);
//   async completions are in-order per wave, so s_wait_asynccnt 4 after
//   issuing the next tile guarantees the current tile's copies landed.
// ---------------------------------------------------------------------------
constexpr int BM = 64, BN = 64, BK = 32, LDSA = 40;

template <int MODE>
__global__ __launch_bounds__(128)
void wmma_gemm_kernel(const half_t* __restrict__ A, const half_t* __restrict__ Bw,
                      const float* __restrict__ bias, const float* __restrict__ scalev,
                      float* __restrict__ Cf, half_t* __restrict__ Ch,
                      int Ndim, int Kdim) {
  __shared__ alignas(16) half_t As[2][BM * LDSA];
  __shared__ alignas(16) half_t Bs[2][BN * LDSA];

  const int tid  = threadIdx.x;
  const int wave = tid >> 5;
  const int lane = tid & 31;
  const int wm = wave >> 1;           // 0..1 -> 32-row group in tile
  const int wn = wave & 1;            // 0..1 -> 32-col group in tile
  const int tileM = blockIdx.x * BM;
  const int tileN = blockIdx.y * BN;

  // per-thread copy geometry: chunks tid and tid+128 of 256 (row 0..63, 8-half seg)
  const int row0 = tid >> 2;
  const int seg0 = (tid & 3) * 8;
  const int row1 = (tid + 128) >> 2;
  const int seg1 = (tid & 3) * 8;     // (tid+128)&3 == tid&3

  const half_t* gA0 = A + (size_t)(tileM + row0) * Kdim + seg0;
  const half_t* gA1 = A + (size_t)(tileM + row1) * Kdim + seg1;
  const half_t* gB0 = Bw + (size_t)(tileN + row0) * Kdim + seg0;
  const half_t* gB1 = Bw + (size_t)(tileN + row1) * Kdim + seg1;

  v8f acc[2][2];
#pragma unroll
  for (int i = 0; i < 2; i++)
#pragma unroll
    for (int j = 0; j < 2; j++)
#pragma unroll
      for (int t = 0; t < 8; t++) acc[i][j][t] = 0.0f;

  const int m_l   = lane & 15;        // row within 16x16 fragment
  const int khalf = (lane >> 4) * 8;  // K offset (+8) for upper lane half

  const int nk = Kdim / BK;

  // prologue: stage tile 0 into buffer 0
  async_copy_b128(&As[0][row0 * LDSA + seg0], gA0);
  async_copy_b128(&As[0][row1 * LDSA + seg1], gA1);
  async_copy_b128(&Bs[0][row0 * LDSA + seg0], gB0);
  async_copy_b128(&Bs[0][row1 * LDSA + seg1], gB1);

  for (int t = 0; t < nk; ++t) {
    const int cur = t & 1;
    if (t + 1 < nk) {
      // stage next tile into the other buffer (its last readers passed the
      // trailing barrier of iteration t-1), then wait for current tile only
      const int koff = (t + 1) * BK;
      async_copy_b128(&As[cur ^ 1][row0 * LDSA + seg0], gA0 + koff);
      async_copy_b128(&As[cur ^ 1][row1 * LDSA + seg1], gA1 + koff);
      async_copy_b128(&Bs[cur ^ 1][row0 * LDSA + seg0], gB0 + koff);
      async_copy_b128(&Bs[cur ^ 1][row1 * LDSA + seg1], gB1 + koff);
      asm volatile("s_wait_asynccnt 0x4" ::: "memory");
    } else {
      asm volatile("s_wait_asynccnt 0x0" ::: "memory");
    }
    __syncthreads();

    // ---- fragment load per documented 16-bit A/B VGPR layout ----
    v16h af[2], bf[2];
#pragma unroll
    for (int i = 0; i < 2; i++) {
      int arow = wm * 32 + i * 16 + m_l;
      int brow = wn * 32 + i * 16 + m_l;
#pragma unroll
      for (int j = 0; j < 16; j++) {
        int kk = ((j & 8) << 1) + (((j >> 1) & 3) << 1) + (j & 1) + khalf;
        af[i][j] = As[cur][arow * LDSA + kk];
        bf[i][j] = Bs[cur][brow * LDSA + kk];
      }
    }
#pragma unroll
    for (int i = 0; i < 2; i++)
#pragma unroll
      for (int j = 0; j < 2; j++)
        acc[i][j] = __builtin_amdgcn_wmma_f32_16x16x32_f16(
            false, af[i], false, bf[j], (short)0, acc[i][j], false, false);
    __syncthreads();   // all waves done reading buf[cur] before it is refilled
  }

  // ---- epilogue; C layout: lane n = lane&15, VGPR v -> m = v + 8*(lane>>4) ----
  const int n_l = lane & 15;
  const int m_hi = (lane >> 4) * 8;
#pragma unroll
  for (int i = 0; i < 2; i++)
#pragma unroll
    for (int j = 0; j < 2; j++) {
      int rbase = tileM + wm * 32 + i * 16 + m_hi;
      int c = tileN + wn * 32 + j * 16 + n_l;
#pragma unroll
      for (int v = 0; v < 8; ++v) {
        int r = rbase + v;
        float val = acc[i][j][v];
        if (MODE == 1) {
          Cf[(size_t)r * Ndim + c] = (val + bias[c]) * scalev[c];
        } else if (MODE == 2) {
          float x = val + bias[c];
          Ch[(size_t)r * Ndim + c] = (half_t)(0.5f * x * (1.0f + erff(x * 0.70710678118654752f)));
        } else {
          Cf[(size_t)r * Ndim + c] = val + bias[c];
        }
      }
    }
}

// ---------------------------------------------------------------------------
// Landmarks: Ql/Kl[b,h,m,d] = mean over 200-seg of Q/K
// ---------------------------------------------------------------------------
__global__ void landmarks_kernel(const float* __restrict__ qkv,
                                 float* __restrict__ ql, float* __restrict__ kl) {
  int idx = blockIdx.x * blockDim.x + threadIdx.x;   // (bh, m, d)
  if (idx >= Bb * Hh * Mm * HD) return;
  int d = idx % HD;
  int m = (idx / HD) % Mm;
  int bh = idx / (HD * Mm);
  int h = bh % Hh;
  int b = bh / Hh;
  float sq = 0.f, sk = 0.f;
  for (int i = 0; i < SM; i++) {
    size_t row = (size_t)(b * Ss + m * SM + i) * QKVN;
    sq += qkv[row + h * HD + d];
    sk += qkv[row + Ee + h * HD + d];
  }
  ql[idx] = sq * (1.0f / SM);
  kl[idx] = sk * (1.0f / SM);
}

// ---------------------------------------------------------------------------
// k1 = softmax_m( Q[b,h,s,:] . Kl[b,h,m,:] )  -- one thread per (b,h,s)
// ---------------------------------------------------------------------------
__global__ void k1_kernel(const float* __restrict__ qkv, const float* __restrict__ kl,
                          float* __restrict__ k1) {
  int idx = blockIdx.x * blockDim.x + threadIdx.x;
  if (idx >= Bb * Hh * Ss) return;
  int s = idx % Ss;
  int bh = idx / Ss;
  int h = bh % Hh;
  int b = bh / Hh;
  const float* qrow = qkv + (size_t)(b * Ss + s) * QKVN + h * HD;
  const float* klm = kl + (size_t)bh * Mm * HD;
  float q[HD];
#pragma unroll
  for (int d = 0; d < HD; d++) q[d] = qrow[d];
  float lg[Mm];
  float mx = -1e30f;
  for (int m = 0; m < Mm; m++) {
    float sdot = 0.f;
    for (int d = 0; d < HD; d++) sdot += q[d] * klm[m * HD + d];
    lg[m] = sdot;
    mx = fmaxf(mx, sdot);
  }
  float ssum = 0.f;
  for (int m = 0; m < Mm; m++) { lg[m] = expf(lg[m] - mx); ssum += lg[m]; }
  float inv = 1.0f / ssum;
  float* out = k1 + (size_t)idx * Mm;
  for (int m = 0; m < Mm; m++) out[m] = lg[m] * inv;
}

// ---------------------------------------------------------------------------
// k2 = softmax_c( Ql[b,h,m,:] . Kl[b,h,c,:] )  -- one thread per (b,h,m)
// ---------------------------------------------------------------------------
__global__ void k2_kernel(const float* __restrict__ ql, const float* __restrict__ kl,
                          float* __restrict__ k2) {
  int idx = blockIdx.x * blockDim.x + threadIdx.x;
  if (idx >= Bb * Hh * Mm) return;
  int mr = idx % Mm;
  int bh = idx / Mm;
  const float* qlr = ql + ((size_t)bh * Mm + mr) * HD;
  const float* klm = kl + (size_t)bh * Mm * HD;
  float lg[Mm];
  float mx = -1e30f;
  for (int c = 0; c < Mm; c++) {
    float sdot = 0.f;
    for (int d = 0; d < HD; d++) sdot += qlr[d] * klm[c * HD + d];
    lg[c] = sdot;
    mx = fmaxf(mx, sdot);
  }
  float ssum = 0.f;
  for (int c = 0; c < Mm; c++) { lg[c] = expf(lg[c] - mx); ssum += lg[c]; }
  float inv = 1.0f / ssum;
  float* out = k2 + (size_t)idx * Mm;
  for (int c = 0; c < Mm; c++) out[c] = lg[c] * inv;
}

// global max |row-sum| / |col-sum| over all (b,h) k2 matrices
__global__ void k2_maxsum_kernel(const float* __restrict__ k2, float* __restrict__ maxv) {
  __shared__ float redc[256];
  __shared__ float redr[256];
  int tid = threadIdx.x;
  float mc = 0.f, mr = 0.f;
  for (int i = tid; i < Bb * Hh * Mm; i += 256) {
    int bh = i / Mm, c = i % Mm;
    const float* mat = k2 + (size_t)bh * Mm * Mm;
    float sc = 0.f, sr = 0.f;
    for (int r2 = 0; r2 < Mm; r2++) {
      sc += fabsf(mat[r2 * Mm + c]);   // column sum (axis=-2)
      sr += fabsf(mat[c * Mm + r2]);   // row sum (axis=-1)
    }
    mc = fmaxf(mc, sc);
    mr = fmaxf(mr, sr);
  }
  redc[tid] = mc; redr[tid] = mr;
  __syncthreads();
  for (int s2 = 128; s2 > 0; s2 >>= 1) {
    if (tid < s2) {
      redc[tid] = fmaxf(redc[tid], redc[tid + s2]);
      redr[tid] = fmaxf(redr[tid], redr[tid + s2]);
    }
    __syncthreads();
  }
  if (tid == 0) { maxv[0] = redc[0]; maxv[1] = redr[0]; }
}

__device__ inline void mm10(const float* a, const float* b, float* o) {
  for (int r = 0; r < Mm; r++)
    for (int c = 0; c < Mm; c++) {
      float s = 0.f;
      for (int k = 0; k < Mm; k++) s += a[r * Mm + k] * b[k * Mm + c];
      o[r * Mm + c] = s;
    }
}

// Newton-Schulz 6-iter pseudo-inverse of each 10x10 k2 -- one thread per (b,h)
__global__ void k2_inv_kernel(const float* __restrict__ k2, const float* __restrict__ maxv,
                              float* __restrict__ k2inv) {
  int bh = blockIdx.x * blockDim.x + threadIdx.x;
  if (bh >= Bb * Hh) return;
  float Km[Mm * Mm], Vm[Mm * Mm], KV[Mm * Mm], T[Mm * Mm], W[Mm * Mm];
  const float* src = k2 + (size_t)bh * Mm * Mm;
  for (int i = 0; i < Mm * Mm; i++) Km[i] = src[i];
  float inv_ab = 1.0f / (maxv[0] * maxv[1]);
  for (int r = 0; r < Mm; r++)
    for (int c = 0; c < Mm; c++) Vm[r * Mm + c] = Km[c * Mm + r] * inv_ab;
  for (int it = 0; it < 6; ++it) {
    mm10(Km, Vm, KV);
    for (int i = 0; i < Mm * Mm; i++) T[i] = ((i % (Mm + 1) == 0) ? 7.0f : 0.0f) - KV[i];
    mm10(KV, T, W);
    for (int i = 0; i < Mm * Mm; i++) T[i] = ((i % (Mm + 1) == 0) ? 15.0f : 0.0f) - W[i];
    mm10(KV, T, W);
    for (int i = 0; i < Mm * Mm; i++) T[i] = ((i % (Mm + 1) == 0) ? 13.0f : 0.0f) - W[i];
    mm10(Vm, T, W);
    for (int i = 0; i < Mm * Mm; i++) Vm[i] = 0.25f * W[i];
  }
  float* dst = k2inv + (size_t)bh * Mm * Mm;
  for (int i = 0; i < Mm * Mm; i++) dst[i] = Vm[i];
}

// ---------------------------------------------------------------------------
// k3 = softmax_s( Ql[b,h,m,:] . K[b,h,s,:] ) -- one block per (b,h,m), row len 2000
// ---------------------------------------------------------------------------
__global__ void k3_kernel(const float* __restrict__ qkv, const float* __restrict__ ql,
                          float* __restrict__ k3) {
  __shared__ float red[256];
  int bhm = blockIdx.x;
  int tid = threadIdx.x;
  int bh = bhm / Mm;
  int h = bh % Hh;
  int b = bh / Hh;
  const float* qlr = ql + (size_t)bhm * HD;
  float qv[HD];
#pragma unroll
  for (int d = 0; d < HD; d++) qv[d] = qlr[d];
  float* row = k3 + (size_t)bhm * Ss;
  float mx = -1e30f;
  for (int s = tid; s < Ss; s += 256) {
    const float* kr = qkv + (size_t)(b * Ss + s) * QKVN + Ee + h * HD;
    float dd = 0.f;
    for (int d = 0; d < HD; d++) dd += qv[d] * kr[d];
    row[s] = dd;
    mx = fmaxf(mx, dd);
  }
  red[tid] = mx;
  __syncthreads();
  for (int s2 = 128; s2 > 0; s2 >>= 1) {
    if (tid < s2) red[tid] = fmaxf(red[tid], red[tid + s2]);
    __syncthreads();
  }
  mx = red[0];
  __syncthreads();
  float ssum = 0.f;
  for (int s = tid; s < Ss; s += 256) {
    float e2 = expf(row[s] - mx);
    row[s] = e2;
    ssum += e2;
  }
  red[tid] = ssum;
  __syncthreads();
  for (int s2 = 128; s2 > 0; s2 >>= 1) {
    if (tid < s2) red[tid] += red[tid + s2];
    __syncthreads();
  }
  float inv = 1.0f / red[0];
  for (int s = tid; s < Ss; s += 256) row[s] *= inv;
}

// ---------------------------------------------------------------------------
// T2 = k2inv @ (k3 @ V)  per (b,h): 10x64 -- one block (640 thr) per (b,h)
// ---------------------------------------------------------------------------
__global__ void t2_kernel(const float* __restrict__ qkv, const float* __restrict__ k3,
                          const float* __restrict__ k2inv, float* __restrict__ t2) {
  __shared__ float T1s[Mm * HD];
  int bh = blockIdx.x;
  int tid = threadIdx.x;                 // 640 threads: (m,d)
  int h = bh % Hh;
  int b = bh / Hh;
  int m = tid / HD;
  int d = tid % HD;
  const float* k3r = k3 + (size_t)(bh * Mm + m) * Ss;
  float acc = 0.f;
  for (int s = 0; s < Ss; s++)
    acc += k3r[s] * qkv[(size_t)(b * Ss + s) * QKVN + 2 * Ee + h * HD + d];
  T1s[tid] = acc;
  __syncthreads();
  const float* ki = k2inv + (size_t)bh * Mm * Mm;
  float acc2 = 0.f;
  for (int j = 0; j < Mm; j++) acc2 += ki[m * Mm + j] * T1s[j * HD + d];
  t2[(size_t)bh * Mm * HD + tid] = acc2;
}

// attn[b,s,h*64+d] = sum_m k1[b,h,s,m] * T2[b,h,m,d]
__global__ void attn_kernel(const float* __restrict__ k1, const float* __restrict__ t2,
                            float* __restrict__ attn) {
  int idx = blockIdx.x * blockDim.x + threadIdx.x;
  if (idx >= Bb * Ss * Ee) return;
  int e = idx & (Ee - 1);
  int row = idx >> 10;
  int h = e >> 6;
  int d = e & 63;
  int b = row / Ss;
  int s = row % Ss;
  int bh = b * Hh + h;
  const float* k1r = k1 + ((size_t)bh * Ss + s) * Mm;
  const float* t2m = t2 + (size_t)bh * Mm * HD;
  float a = 0.f;
  for (int m = 0; m < Mm; m++) a += k1r[m] * t2m[m * HD + d];
  attn[idx] = a;
}

// ---------------------------------------------------------------------------
// LayerNorm over E=1024. OUTMODE 0: write f16 (MLP input). OUTMODE 1: write f32
// transposed to (B,E,S) -> final output.
// ---------------------------------------------------------------------------
template <int OUTMODE>
__global__ void layernorm_kernel(const float* __restrict__ in, const float* __restrict__ g,
                                 const float* __restrict__ be, half_t* __restrict__ outh,
                                 float* __restrict__ outf) {
  __shared__ float red[256];
  int row = blockIdx.x;
  int tid = threadIdx.x;
  const float* xr = in + (size_t)row * Ee;
  float s = 0.f;
  for (int e = tid; e < Ee; e += 256) s += xr[e];
  red[tid] = s;
  __syncthreads();
  for (int s2 = 128; s2 > 0; s2 >>= 1) {
    if (tid < s2) red[tid] += red[tid + s2];
    __syncthreads();
  }
  float mu = red[0] * (1.0f / Ee);
  __syncthreads();
  float v = 0.f;
  for (int e = tid; e < Ee; e += 256) {
    float d2 = xr[e] - mu;
    v += d2 * d2;
  }
  red[tid] = v;
  __syncthreads();
  for (int s2 = 128; s2 > 0; s2 >>= 1) {
    if (tid < s2) red[tid] += red[tid + s2];
    __syncthreads();
  }
  float rstd = rsqrtf(red[0] * (1.0f / Ee) + LNEPS);
  for (int e = tid; e < Ee; e += 256) {
    float val = (xr[e] - mu) * rstd * g[e] + be[e];
    if (OUTMODE == 0) {
      outh[(size_t)row * Ee + e] = (half_t)val;
    } else {
      int b = row / Ss;
      int s_ = row % Ss;
      outf[((size_t)b * Ee + e) * Ss + s_] = val;
    }
  }
}

// ---------------------------------------------------------------------------
// Launcher
// ---------------------------------------------------------------------------
extern "C" void kernel_launch(void* const* d_in, const int* in_sizes, int n_in,
                              void* d_out, int out_size, void* d_ws, size_t ws_size,
                              hipStream_t stream) {
  (void)in_sizes; (void)n_in; (void)out_size; (void)ws_size;

  const float* X   = (const float*)d_in[0];
  const float* Wq  = (const float*)d_in[1];
  const float* bq  = (const float*)d_in[2];
  const float* Wk  = (const float*)d_in[3];
  const float* bk  = (const float*)d_in[4];
  const float* Wv  = (const float*)d_in[5];
  const float* bvv = (const float*)d_in[6];
  const float* g1  = (const float*)d_in[7];
  const float* be1 = (const float*)d_in[8];
  const float* W1  = (const float*)d_in[9];
  const float* b1  = (const float*)d_in[10];
  const float* W2  = (const float*)d_in[11];
  const float* b2  = (const float*)d_in[12];
  const float* g2  = (const float*)d_in[13];
  const float* be2 = (const float*)d_in[14];
  float* out = (float*)d_out;

  char* base = (char*)d_ws;
  size_t off = 0;
  auto alloc = [&](size_t bytes) -> char* {
    char* p = base + off;
    off += (bytes + 255) & ~(size_t)255;
    return p;
  };

  half_t* xh     = (half_t*)alloc((size_t)Bb * Ss * Ee * 2);
  half_t* wqkv   = (half_t*)alloc((size_t)QKVN * Ee * 2);
  half_t* w1h    = (half_t*)alloc((size_t)MLPd * Ee * 2);
  half_t* w2h    = (half_t*)alloc((size_t)Ee * MLPd * 2);
  float*  biasc  = (float*)alloc((size_t)QKVN * 4);
  float*  scalec = (float*)alloc((size_t)QKVN * 4);
  float*  qkv    = (float*)alloc((size_t)Bb * Ss * QKVN * 4);
  float*  ql     = (float*)alloc((size_t)Bb * Hh * Mm * HD * 4);
  float*  kl     = (float*)alloc((size_t)Bb * Hh * Mm * HD * 4);
  float*  k1     = (float*)alloc((size_t)Bb * Hh * Ss * Mm * 4);
  float*  k2     = (float*)alloc((size_t)Bb * Hh * Mm * Mm * 4);
  float*  k2inv  = (float*)alloc((size_t)Bb * Hh * Mm * Mm * 4);
  float*  maxv   = (float*)alloc(2 * 4);
  float*  k3     = (float*)alloc((size_t)Bb * Hh * Mm * Ss * 4);
  float*  t2     = (float*)alloc((size_t)Bb * Hh * Mm * HD * 4);
  float*  attn   = (float*)alloc((size_t)Bb * Ss * Ee * 4);
  half_t* hh     = (half_t*)alloc((size_t)Bb * Ss * Ee * 2);
  half_t* mid    = (half_t*)alloc((size_t)Bb * Ss * MLPd * 2);
  float*  out2   = (float*)alloc((size_t)Bb * Ss * Ee * 4);

  const int ROWS = Bb * Ss;  // 16000

  // 1. inputs -> f16
  xpose_to_f16_kernel<<<(Bb * Ss * Ee + 255) / 256, 256, 0, stream>>>(X, xh);
  cvt_f16_kernel<<<(Hh * HD * Ee + 255) / 256, 256, 0, stream>>>(Wq, wqkv, Hh * HD * Ee);
  cvt_f16_kernel<<<(Hh * HD * Ee + 255) / 256, 256, 0, stream>>>(Wk, wqkv + (size_t)Ee * Ee, Hh * HD * Ee);
  cvt_f16_kernel<<<(Hh * HD * Ee + 255) / 256, 256, 0, stream>>>(Wv, wqkv + (size_t)2 * Ee * Ee, Hh * HD * Ee);
  cvt_f16_kernel<<<(MLPd * Ee + 255) / 256, 256, 0, stream>>>(W1, w1h, MLPd * Ee);
  cvt_f16_kernel<<<(Ee * MLPd + 255) / 256, 256, 0, stream>>>(W2, w2h, Ee * MLPd);
  build_bias_scale_kernel<<<(QKVN + 255) / 256, 256, 0, stream>>>(bq, bk, bvv, biasc, scalec);

  // 2. fused QKV projection (+bias, +scale on Q/K): (16000x1024)x(3072x1024)^T
  wmma_gemm_kernel<1><<<dim3(ROWS / BM, QKVN / BN), 128, 0, stream>>>(
      xh, wqkv, biasc, scalec, qkv, nullptr, QKVN, Ee);

  // 3. attention pieces (all tiny vs the GEMMs)
  landmarks_kernel<<<(Bb * Hh * Mm * HD + 255) / 256, 256, 0, stream>>>(qkv, ql, kl);
  k1_kernel<<<(Bb * Hh * Ss + 255) / 256, 256, 0, stream>>>(qkv, kl, k1);
  k2_kernel<<<(Bb * Hh * Mm + 255) / 256, 256, 0, stream>>>(ql, kl, k2);
  k2_maxsum_kernel<<<1, 256, 0, stream>>>(k2, maxv);
  k2_inv_kernel<<<1, 128, 0, stream>>>(k2, maxv, k2inv);
  k3_kernel<<<Bb * Hh * Mm, 256, 0, stream>>>(qkv, ql, k3);
  t2_kernel<<<Bb * Hh, Mm * HD, 0, stream>>>(qkv, k3, k2inv, t2);
  attn_kernel<<<(Bb * Ss * Ee + 255) / 256, 256, 0, stream>>>(k1, t2, attn);

  // 4. LN1 -> f16, MLP up (GELU) -> f16, MLP down -> f32, LN2 -> transposed out
  layernorm_kernel<0><<<ROWS, 256, 0, stream>>>(attn, g1, be1, hh, nullptr);
  wmma_gemm_kernel<2><<<dim3(ROWS / BM, MLPd / BN), 128, 0, stream>>>(
      hh, w1h, b1, nullptr, nullptr, mid, MLPd, Ee);
  wmma_gemm_kernel<3><<<dim3(ROWS / BM, Ee / BN), 128, 0, stream>>>(
      mid, w2h, b2, nullptr, out2, nullptr, Ee, MLPd);
  layernorm_kernel<1><<<ROWS, 256, 0, stream>>>(out2, g2, be2, nullptr, out);
}
